// LSTM_3238405341527
// MI455X (gfx1250) — matrix operand order repaired and checked by reading, not compiled, not run on hardware
//
#include <hip/hip_runtime.h>
#include <stdint.h>

// ---------------------------------------------------------------------------
// LSTM on MI455X (gfx1250), bf16 WMMA with fp32 accumulate.
// N=64, T=1024, D=1024, H=1024.
//
// Workspace layout (requires ~1.22 GiB):
//   xW      : [65536][4096] f32   (x@Wx + b, row = n*T + t)        1 GiB
//   x_pack  : A-fragment bf16 [4096 mtiles][32 ksteps][32 lanes][16] 128 MiB
//   Wx_pack : B-fragment bf16 [256 ncbs][32 ksteps][32 lanes][16]    8 MiB
//   Wh_pack : same                                                   8 MiB
//   h_pack0/1: A-fragment bf16 [4][32][32][16] (double buffered)   2*128 KiB
//   c state : [64][1024] f32                                       256 KiB
// ---------------------------------------------------------------------------

typedef __attribute__((ext_vector_type(16))) __bf16   bf16x16;
typedef __attribute__((ext_vector_type(8)))  float    f32x8;
typedef __attribute__((ext_vector_type(4)))  uint32_t u32x4;

__device__ inline uint32_t f2bf(float f) {
  uint32_t u = __float_as_uint(f);
  return (u + 0x7FFFu + ((u >> 16) & 1u)) >> 16;   // round-to-nearest-even
}
__device__ inline uint32_t pack2bf(float lo, float hi) {
  return f2bf(lo) | (f2bf(hi) << 16);
}

__device__ inline bf16x16 load_frag(const u32x4* __restrict__ p) {
  union U { bf16x16 v; u32x4 q[2]; } u;
  u.q[0] = p[0];
  u.q[1] = p[1];
  return u.v;
}

__device__ inline void store8(uint32_t* __restrict__ dst, const uint32_t o[8]) {
  u32x4 a = { o[0], o[1], o[2], o[3] };
  u32x4 b = { o[4], o[5], o[6], o[7] };
  ((u32x4*)dst)[0] = a;
  ((u32x4*)dst)[1] = b;
}

// CDNA5 async copy: 16B global -> LDS per lane, tracked by ASYNCcnt.
// (ISA 15.18.3 GLOBAL_LOAD_ASYNC_TO_LDS_B128; VDST = LDS byte address VGPR.)
__device__ inline void async_copy_b128(uint32_t lds_off, const void* gaddr) {
  asm volatile("global_load_async_to_lds_b128 %0, %1, off"
               :: "v"(lds_off), "v"(gaddr) : "memory");
}
__device__ inline void wait_asynccnt0() {
  asm volatile("s_wait_asynccnt 0x0" ::: "memory");
}

#define WMMA_BF16(A, B, Cacc) \
  __builtin_amdgcn_wmma_f32_16x16x32_bf16(false, (A), false, (B), (short)0, (Cacc), false, false)

// ---------------------------------------------------------------------------
// Pack fp32 row-major [M][ldk] matrix into WMMA A-fragment bf16 layout:
//   dst[((mtile*ksteps + kstep)*32 + lane)*8 u32]
// A 16x32 bf16 layout (ISA 7.12.2): lane = (m%16) + 16*lh; vgpr i holds
//   K = (i>=4?16:0) + lh*8 + (i&3)*2 + {0,1}
// ---------------------------------------------------------------------------
__global__ void pack_a_kernel(const float* __restrict__ src, uint32_t* __restrict__ dst,
                              int mtiles, int ksteps, int ldk) {
  int tid = blockIdx.x * blockDim.x + threadIdx.x;
  if (tid >= mtiles * ksteps * 32) return;
  int lane  = tid & 31;
  int frag  = tid >> 5;
  int kstep = frag % ksteps;
  int mtile = frag / ksteps;
  int m  = mtile * 16 + (lane & 15);
  int lh = lane >> 4;
  const float* row = src + (size_t)m * ldk + kstep * 32;
  uint32_t o[8];
#pragma unroll
  for (int i = 0; i < 8; ++i) {
    int k = ((i >= 4) ? 16 : 0) + lh * 8 + (i & 3) * 2;
    o[i] = pack2bf(row[k], row[k + 1]);
  }
  store8(dst + (size_t)tid * 8, o);
}

// ---------------------------------------------------------------------------
// Pack fp32 row-major weight [K][ldn] into WMMA B-fragment bf16 layout:
//   dst[((ncb*ksteps + kstep)*32 + lane)*8 u32]
// B 32x16 bf16 layout: lane = (n%16) + 16*lh; vgpr i holds K = (i>=4?16:0) + lh*8 + (i&3)*2 + {0,1}
// ---------------------------------------------------------------------------
__global__ void pack_b_kernel(const float* __restrict__ W, uint32_t* __restrict__ dst,
                              int ncbs, int ksteps, int ldn) {
  int tid = blockIdx.x * blockDim.x + threadIdx.x;
  if (tid >= ncbs * ksteps * 32) return;
  int lane  = tid & 31;
  int frag  = tid >> 5;
  int kstep = frag % ksteps;
  int ncb   = frag / ksteps;
  int n  = ncb * 16 + (lane & 15);
  int k0 = kstep * 32 + (lane >> 4) * 8;
  uint32_t o[8];
#pragma unroll
  for (int i = 0; i < 8; ++i) {
    int k = k0 + ((i >= 4) ? 16 : 0) + (i & 3) * 2;
    o[i] = pack2bf(W[(size_t)k * ldn + n], W[(size_t)(k + 1) * ldn + n]);
  }
  store8(dst + (size_t)tid * 8, o);
}

__global__ void zero_f32_kernel(float* __restrict__ p, int n) {
  int i = blockIdx.x * blockDim.x + threadIdx.x;
  if (i < n) p[i] = 0.0f;
}

// ---------------------------------------------------------------------------
// Phase 1: xW = x @ Wx + b.   M=65536, N=4096, K=1024, bf16 WMMA, f32 acc.
// Grid: (32, 512) x 256 threads (8 waves). Wave w owns mtile = blockIdx.y*8+w;
// all 8 waves share 8 N-blocks (128 cols), so B is staged once per WG into
// LDS via the CDNA5 async-to-LDS path. Each stage covers 2 k-steps (16 KB,
// double buffered, 4x16B async chunks per thread), so one workgroup barrier
// amortizes over 16 WMMAs per wave. A fragments are register-pipelined.
// ---------------------------------------------------------------------------
__global__ __launch_bounds__(256)
void gemm_xw_kernel(const u32x4* __restrict__ Apack, const u32x4* __restrict__ Bpack,
                    const float* __restrict__ bias, float* __restrict__ xW) {
  // [buf][frag(kk*8+j)][lane][half] : 16 KB per buffer, 32 KB total.
  __shared__ u32x4 lds_b[2][16 * 64];

  const int tid   = threadIdx.x;
  const int lane  = tid & 31;
  const int wave  = tid >> 5;
  const int mtile = blockIdx.y * 8 + wave;   // 0..4095
  const int ncb0  = blockIdx.x * 8;          // 0..248

  // Async staging: 4 chunks of 16B per thread per stage (1024 chunks total).
  // chunk c = tid + 256*q: frag f = c>>6 (= kk*8 + j), lane cl, half ch.
  const u32x4* bsrc[4];
  int lidx[4];
#pragma unroll
  for (int q = 0; q < 4; ++q) {
    int c  = tid + 256 * q;
    int f  = c >> 6;
    int kk = f >> 3, jj = f & 7;
    int cl = (c >> 1) & 31, ch = c & 1;
    bsrc[q] = Bpack + (((size_t)(ncb0 + jj) * 32 + kk) * 32 + cl) * 2 + ch; // +s*128
    lidx[q] = f * 64 + cl * 2 + ch;
  }

  auto stage_in = [&](int buf, int s) {
#pragma unroll
    for (int q = 0; q < 4; ++q)
      async_copy_b128((uint32_t)(uintptr_t)&lds_b[buf][lidx[q]],
                      bsrc[q] + (size_t)s * 128);
  };

  stage_in(0, 0);
  const u32x4* aptr = Apack + ((size_t)mtile * 32 * 32 + lane) * 2;  // +ks*64
  bf16x16 a0 = load_frag(aptr);            // ks = 0
  bf16x16 a1 = load_frag(aptr + 64);       // ks = 1
  wait_asynccnt0();
  __syncthreads();

  f32x8 C[8] = {};
  for (int s = 0; s < 16; ++s) {
    const int cur = s & 1;
    bf16x16 aN0, aN1;
    if (s < 15) {
      stage_in(cur ^ 1, s + 1);
      aN0 = load_frag(aptr + (size_t)(2 * s + 2) * 64);
      aN1 = load_frag(aptr + (size_t)(2 * s + 3) * 64);
    }
    const u32x4* lb = &lds_b[cur][0];
#pragma unroll
    for (int kk = 0; kk < 2; ++kk) {
      bf16x16 a = kk ? a1 : a0;
#pragma unroll
      for (int jh = 0; jh < 2; ++jh) {     // two groups of 4 N-blocks
        bf16x16 bL[4];
#pragma unroll
        for (int j = 0; j < 4; ++j)
          bL[j] = load_frag(lb + (kk * 8 + jh * 4 + j) * 64 + lane * 2);
#pragma unroll
        for (int j = 0; j < 4; ++j)
          C[jh * 4 + j] = WMMA_BF16(a, bL[j], C[jh * 4 + j]);
      }
    }
    if (s < 15) {
      a0 = aN0;
      a1 = aN1;
      wait_asynccnt0();
      __syncthreads();
    }
  }

  const int rowadd = 8 * (lane >> 4);
#pragma unroll
  for (int j = 0; j < 8; ++j) {
    int col  = (ncb0 + j) * 16 + (lane & 15);
    float bb = bias[col];
#pragma unroll
    for (int r = 0; r < 8; ++r) {
      int m = mtile * 16 + r + rowadd;
      xW[(size_t)m * 4096 + col] = C[j][r] + bb;   // row m = n*T + t
    }
  }
}

// ---------------------------------------------------------------------------
// Recurrent step: preact = xW[:,t,:] + h@Wh; gates; update c; emit h.
// Grid: 32 WGs x 256 threads (8 waves). WG owns 32 h-columns [cb*32, +32) ==
// exactly A-fragment kstep `cb` of the next step (local repack via LDS).
// Wave w: mt = w&3 (16 rows), half = w>>2 (16 cols), all 4 gates.
// K-loop is software-pipelined with two full register sets (even/odd) so the
// fragment loads of step k+1/k+2 overlap the WMMAs of step k.
// h_pack is double buffered across launches (cross-WG RAW on h).
// ---------------------------------------------------------------------------
__global__ __launch_bounds__(256)
void lstm_step_kernel(const u32x4* __restrict__ hpack_in, uint32_t* __restrict__ hpack_out,
                      const u32x4* __restrict__ Whpack, const float* __restrict__ xW,
                      float* __restrict__ cstate, float* __restrict__ out, int t) {
  __shared__ unsigned short lds_h[4 * 16 * 32];   // [mt][row16][col32] bf16 bits

  const int lane = threadIdx.x & 31;
  const int wave = threadIdx.x >> 5;
  const int mt   = wave & 3;
  const int half = wave >> 2;
  const int cb   = blockIdx.x;                    // 0..31

  const u32x4* aptr = hpack_in + ((size_t)mt * 32 * 32 + lane) * 2;  // +ks*64
  const u32x4* bptr[4];
#pragma unroll
  for (int g = 0; g < 4; ++g) {
    int ncb = g * 64 + cb * 2 + half;
    bptr[g] = Whpack + ((size_t)ncb * 32 * 32 + lane) * 2;           // +ks*64
  }

  // Even/odd register sets, prefetch distance 1-2 k-steps.
  bf16x16 aE = load_frag(aptr);
  bf16x16 bE[4], bO[4];
#pragma unroll
  for (int g = 0; g < 4; ++g) bE[g] = load_frag(bptr[g]);

  f32x8 C[4] = {};
  for (int ks = 0; ks < 32; ks += 2) {
    bf16x16 aO = load_frag(aptr + (size_t)(ks + 1) * 64);
#pragma unroll
    for (int g = 0; g < 4; ++g) bO[g] = load_frag(bptr[g] + (size_t)(ks + 1) * 64);
#pragma unroll
    for (int g = 0; g < 4; ++g) C[g] = WMMA_BF16(aE, bE[g], C[g]);
    if (ks < 30) {
      aE = load_frag(aptr + (size_t)(ks + 2) * 64);
#pragma unroll
      for (int g = 0; g < 4; ++g) bE[g] = load_frag(bptr[g] + (size_t)(ks + 2) * 64);
    }
#pragma unroll
    for (int g = 0; g < 4; ++g) C[g] = WMMA_BF16(aO, bO[g], C[g]);
  }

  const int j      = cb * 32 + half * 16 + (lane & 15);   // h column 0..1023
  const int rowadd = 8 * (lane >> 4);
#pragma unroll
  for (int r = 0; r < 8; ++r) {
    int n = mt * 16 + r + rowadd;                          // batch row
    const float* xwrow = xW + ((size_t)(n * 1024 + t)) * 4096;
    float ai = C[0][r] + xwrow[j];
    float af = C[1][r] + xwrow[1024 + j];
    float ao = C[2][r] + xwrow[2048 + j];
    float ag = C[3][r] + xwrow[3072 + j];
    float ig = 1.0f / (1.0f + __expf(-ai));
    float fg = 1.0f / (1.0f + __expf(-af));
    float og = 1.0f / (1.0f + __expf(-ao));
    float gg = tanhf(ag);
    size_t ci = (size_t)n * 1024 + j;
    float cn = fg * cstate[ci] + ig * gg;
    cstate[ci] = cn;
    float hn = og * tanhf(cn);
    out[(size_t)n * (1024 * 1024) + (size_t)t * 1024 + j] = hn;
    lds_h[(mt * 16 + r + rowadd) * 32 + (half * 16 + (lane & 15))] =
        (unsigned short)f2bf(hn);
  }

  __syncthreads();

  // Repack this WG's 32 h-columns into the A-fragment for kstep == cb.
  if (wave < 4) {
    int m = lane & 15, lh = lane >> 4;
    uint32_t o[8];
#pragma unroll
    for (int i = 0; i < 8; ++i) {
      int kk = ((i >= 4) ? 16 : 0) + lh * 8 + (i & 3) * 2;
      uint32_t lo = lds_h[(wave * 16 + m) * 32 + kk];
      uint32_t hi = lds_h[(wave * 16 + m) * 32 + kk + 1];
      o[i] = lo | (hi << 16);
    }
    store8(hpack_out + ((size_t)(wave * 32 + cb) * 32 + lane) * 8, o);
  }
}

// ---------------------------------------------------------------------------
extern "C" void kernel_launch(void* const* d_in, const int* in_sizes, int n_in,
                              void* d_out, int out_size, void* d_ws, size_t ws_size,
                              hipStream_t stream) {
  const float* x  = (const float*)d_in[0];   // [64][1024][1024]
  const float* h0 = (const float*)d_in[1];   // [64][1024]
  const float* Wx = (const float*)d_in[2];   // [1024][4096]
  const float* Wh = (const float*)d_in[3];   // [1024][4096]
  const float* b  = (const float*)d_in[4];   // [4096]
  float* out = (float*)d_out;                // [64][1024][1024]

  char* ws = (char*)d_ws;
  size_t off = 0;
  float*    xW     = (float*)(ws + off);    off += (size_t)65536 * 4096 * 4;   // 1 GiB
  uint32_t* xpack  = (uint32_t*)(ws + off); off += (size_t)65536 * 1024 * 2;   // 128 MiB
  uint32_t* wxpack = (uint32_t*)(ws + off); off += (size_t)256 * 32 * 32 * 32; // 8 MiB
  uint32_t* whpack = (uint32_t*)(ws + off); off += (size_t)256 * 32 * 32 * 32; // 8 MiB
  uint32_t* hpack0 = (uint32_t*)(ws + off); off += (size_t)4 * 32 * 32 * 32;   // 128 KiB
  uint32_t* hpack1 = (uint32_t*)(ws + off); off += (size_t)4 * 32 * 32 * 32;   // 128 KiB
  float*    cstate = (float*)(ws + off);    off += (size_t)64 * 1024 * 4;      // 256 KiB

  // Pack activations + weights to bf16 WMMA fragment layouts.
  pack_a_kernel<<<(4096 * 32 * 32) / 256, 256, 0, stream>>>(x, xpack, 4096, 32, 1024);
  pack_b_kernel<<<(256 * 32 * 32) / 256, 256, 0, stream>>>(Wx, wxpack, 256, 32, 4096);
  pack_b_kernel<<<(256 * 32 * 32) / 256, 256, 0, stream>>>(Wh, whpack, 256, 32, 4096);
  pack_a_kernel<<<(4 * 32 * 32 + 255) / 256, 256, 0, stream>>>(h0, hpack0, 4, 32, 1024);
  zero_f32_kernel<<<(65536) / 256, 256, 0, stream>>>(cstate, 65536);

  // Phase 1: big input-projection GEMM.
  gemm_xw_kernel<<<dim3(32, 512), 256, 0, stream>>>(
      (const u32x4*)xpack, (const u32x4*)wxpack, b, xW);

  // Phase 2: sequential recurrence; launch boundary == grid-wide sync on h.
  for (int t = 0; t < 1024; ++t) {
    const u32x4* hin = (const u32x4*)((t & 1) ? hpack1 : hpack0);
    uint32_t*   hout = (t & 1) ? hpack0 : hpack1;
    lstm_step_kernel<<<32, 256, 0, stream>>>(
        hin, hout, (const u32x4*)whpack, xW, cstate, out, t);
  }
}
